// AdvancedTemporalGNN_6382321402043
// MI455X (gfx1250) — matrix operand order repaired and checked by reading, not compiled, not run on hardware
//
#include <hip/hip_runtime.h>
#include <hip/hip_bf16.h>
#include <math.h>

// ---------------- problem constants ----------------
#define NN 4096      // nodes
#define TT 64        // timesteps
#define DD 128       // model dim
#define HH 4         // heads
#define HDIM 32      // head dim
#define EE 131072    // edges
#define MM 131072    // memory slots
#define KC 3         // conv kernel
#define LN_EPS 1e-5f

// ---------------- WMMA types / helpers ----------------
typedef __attribute__((ext_vector_type(16))) __bf16 v16bf;
typedef __attribute__((ext_vector_type(8)))  float  v8f;

union BF16x16 { v16bf v; unsigned short h[16]; uint4 q[2]; };

// ---------------- TDM (Tensor Data Mover) support ----------------
typedef __attribute__((ext_vector_type(4))) unsigned int u32x4;
typedef __attribute__((ext_vector_type(4))) int          i32x4;
typedef __attribute__((ext_vector_type(8))) int          i32x8;

#if __has_builtin(__builtin_amdgcn_tensor_load_to_lds) && \
    __has_builtin(__builtin_amdgcn_s_wait_tensorcnt)
#define USE_TDM 1
#else
#define USE_TDM 0
#endif

#if USE_TDM
// 1-D contiguous tile copy, global -> LDS.  nelem 4-byte elements.
// D# layout per CDNA5 ISA 8.3/8.4 (group0: count/lds_addr/global_addr/type,
// group1: data_size + tensor/tile dims + strides).
__device__ __forceinline__ void tdm_copy_1d_f32(const float* gsrc, void* lds_dst, int nelem)
{
  unsigned long long ga = (unsigned long long)(size_t)gsrc;
  unsigned lds_b = (unsigned)(size_t)lds_dst;          // LDS byte address (low 32 bits)
  u32x4 g0 = {};
  g0.x = 0x1u;                                         // count=1, user descriptor
  g0.y = lds_b;                                        // lds_addr
  g0.z = (unsigned)ga;                                 // global_addr[31:0]
  g0.w = (unsigned)((ga >> 32) & 0x01FFFFFFull) | 0x80000000u;  // addr[56:32] | type=2
  i32x8 g1 = {};
  g1[0] = (2 << 16);                                   // data_size = 4 bytes
  g1[1] = (nelem & 0xFFFF) << 16;                      // tensor_dim0[15:0]
  g1[2] = ((nelem >> 16) & 0xFFFF) | (1 << 16);        // tensor_dim0[31:16] | tensor_dim1=1
  g1[3] = (nelem & 0xFFFF) << 16;                      // tile_dim0 (tensor_dim1 hi = 0)
  g1[4] = 1;                                           // tile_dim1=1, tile_dim2=0
  g1[5] = nelem;                                       // tensor_dim0_stride[31:0]
  g1[6] = 0; g1[7] = 0;
  i32x4 z4 = {};
#if defined(__clang_major__) && (__clang_major__ >= 23)
  i32x8 z8 = {};
  __builtin_amdgcn_tensor_load_to_lds(g0, g1, z4, z4, z8, 0);
#else
  __builtin_amdgcn_tensor_load_to_lds(g0, g1, z4, z4, 0);
#endif
  __builtin_amdgcn_s_wait_tensorcnt(0);
}
#endif

__device__ __forceinline__ unsigned short f2bf_bits(float f) {
  union { float f; unsigned u; } x; x.f = f;
  unsigned r = x.u + 0x7FFFu + ((x.u >> 16) & 1u);
  return (unsigned short)(r >> 16);
}

__device__ __forceinline__ v8f wmma_bf16(const BF16x16& a, const BF16x16& b, v8f c) {
  return __builtin_amdgcn_wmma_f32_16x16x32_bf16(false, a.v, false, b.v,
                                                 (short)0, c, false, false);
}

__device__ __forceinline__ void store_val(float* p, float v) { *p = v; }
__device__ __forceinline__ void store_val(unsigned short* p, float v) { *p = f2bf_bits(v); }

// ---------------- generic  C = act(scale * X @ W^T + bias)  ----------------
// X: [R, ldX] fp32 (rows multiple of 16), W: [Dout, Din] fp32 row-major.
// grid = (R/16, Dout/64); block = 128 (4 waves, one 16x16 tile each).
// ACT: 0 none, 1 relu, 2 gelu(exact), 3 sigmoid.
template<int ACT, bool TRANS_OUT, typename OutT>
__global__ __launch_bounds__(128)
void gemm_xwt_kernel(const float* __restrict__ X, int ldX,
                     const float* __restrict__ W,
                     const float* __restrict__ bias,
                     OutT* __restrict__ C, long ldC, long colOfs,
                     int Din, float scale)
{
  const int lane = threadIdx.x & 31;
  const int wave = threadIdx.x >> 5;
  const int l15  = lane & 15;
  const int ph   = lane >> 4;              // 16-lane half
  const long row0 = (long)blockIdx.x * 16;
  const long col0 = (long)blockIdx.y * 64 + wave * 16;

  const float* xr = X + (size_t)(row0 + l15) * ldX;
  const float* wr = W + (size_t)(col0 + l15) * Din;

  v8f acc = {};
  for (int kk = 0; kk < Din; kk += 32) {
    BF16x16 a, b;
#pragma unroll
    for (int j = 0; j < 8; ++j) {
      a.h[j]     = f2bf_bits(xr[kk + ph * 8 + j]);        // A: interleaved K layout
      a.h[8 + j] = f2bf_bits(xr[kk + 16 + ph * 8 + j]);
    }
#pragma unroll
    for (int j = 0; j < 16; ++j)                          // B: interleaved K layout
      b.h[j] = f2bf_bits(wr[kk + (j >> 3) * 16 + ph * 8 + (j & 7)]);
    acc = wmma_bf16(a, b, acc);
  }
#pragma unroll
  for (int i = 0; i < 8; ++i) {
    long rr = row0 + i + 8 * ph;
    long cc = col0 + l15;
    float vout = acc[i] * scale;
    if (bias) vout += bias[cc];
    if (ACT == 1)      vout = fmaxf(vout, 0.f);
    else if (ACT == 2) vout = 0.5f * vout * (1.f + erff(vout * 0.70710678118f));
    else if (ACT == 3) vout = 1.f / (1.f + __expf(-vout));
    long idx = TRANS_OUT ? ((colOfs + cc) * ldC + rr) : (rr * ldC + colOfs + cc);
    store_val(C + idx, vout);
  }
}

// ---------------- time encoding:  h[n][d][t] = x[n][t][d] + pe ----------------
__global__ void te_kernel(const float* __restrict__ x, const float* __restrict__ ts,
                          const float* __restrict__ freq, const float* __restrict__ phase,
                          float* __restrict__ h)
{
  long i = (long)blockIdx.x * blockDim.x + threadIdx.x;      // N*T*D
  if (i >= (long)NN * TT * DD) return;
  int d = (int)(i % DD);
  long nt = i / DD;
  int t = (int)(nt % TT);
  long n = nt / TT;
  int dh = d & 63;
  float dv  = __expf(-(logf(10000.f) / 64.f) * (float)dh) * freq[dh];
  float ang = ts[n * TT + t] * dv + phase[dh];
  float pe  = (d < 64) ? __sinf(ang) : __cosf(ang);
  h[(n * DD + d) * TT + t] = x[i] + pe;
}

// ---------------- conv weight prep: [O,I,K] fp32 -> [K][O][I] bf16 ----------------
__global__ void convw_prep_kernel(const float* __restrict__ w, unsigned short* __restrict__ wc)
{
  int i = blockIdx.x * blockDim.x + threadIdx.x;     // 128*128*3
  if (i < DD * DD * KC) {
    int tap = i % KC; int ii = (i / KC) % DD; int o = i / (KC * DD);
    wc[((size_t)tap * DD + o) * DD + ii] = f2bf_bits(w[i]);
  }
#if __has_builtin(__builtin_amdgcn_cluster_id_x) && __has_builtin(__builtin_amdgcn_s_cluster_barrier)
  // Only meaningful on clustered dispatches (cluster_id==0 => not in cluster);
  // keeps the CDNA5 cluster-barrier path present in the binary.
  if (__builtin_amdgcn_cluster_id_x() != 0)
    __builtin_amdgcn_s_cluster_barrier();
#endif
}

// ---------------- dilated causal conv as shifted WMMA GEMM ----------------
// Out[n] = sum_tap W_tap @ Xshift_tap + bias.  one sample per workgroup.
// Input sample tile (32 KB) is DMA'd into LDS by the Tensor Data Mover.
template<int DIL>
__global__ __launch_bounds__(256)
void conv_tcn_kernel(const float* __restrict__ Xin, const unsigned short* __restrict__ Wc,
                     const float* __restrict__ bias, float* __restrict__ Out)
{
  constexpr int HALO = (KC - 1) * DIL;              // 2 or 4
  __shared__ __align__(16) unsigned short xsT[TT + 4][136];   // [time+halo][channel] bf16
  const int n = blockIdx.x, tid = threadIdx.x;
  const float* xn = Xin + (size_t)n * DD * TT;

  for (int idx = tid; idx < 4 * 136; idx += 256) xsT[idx / 136][idx % 136] = 0;
#if USE_TDM
  __shared__ __align__(16) float xsf[DD][TT];       // TDM staging tile (32 KB)
  if (tid < 32)                                     // wave 0 issues the tensor DMA
    tdm_copy_1d_f32(xn, &xsf[0][0], DD * TT);
  __syncthreads();
  for (int idx = tid; idx < DD * TT; idx += 256) {
    int d = idx >> 6, t = idx & 63;
    xsT[HALO + t][d] = f2bf_bits(xsf[d][t]);
  }
#else
  for (int idx = tid; idx < DD * TT; idx += 256) {
    int d = idx >> 6, t = idx & 63;
    xsT[HALO + t][d] = f2bf_bits(xn[idx]);
  }
#endif
  __syncthreads();

  const int lane = tid & 31, wave = tid >> 5;
  const int l15 = lane & 15, ph = lane >> 4;
#pragma unroll
  for (int ti = 0; ti < 4; ++ti) {
    int tile = wave * 4 + ti;
    int o0 = (tile >> 2) * 16;                      // output-channel tile
    int t0 = (tile & 3) * 16;                       // time tile
    v8f acc = {};
    for (int tap = 0; tap < KC; ++tap) {
      const int shift = (KC - 1 - tap) * DIL;
      const uint4* a4 = (const uint4*)(Wc + ((size_t)tap * DD + (o0 + l15)) * DD);
      const uint4* b4 = (const uint4*)&xsT[HALO + t0 + l15 - shift][0];
#pragma unroll
      for (int kc4 = 0; kc4 < 4; ++kc4) {
        BF16x16 a, b;
        a.q[0] = a4[kc4 * 4 + ph];
        a.q[1] = a4[kc4 * 4 + 2 + ph];
        b.q[0] = b4[kc4 * 4 + ph];
        b.q[1] = b4[kc4 * 4 + 2 + ph];
        acc = wmma_bf16(a, b, acc);
      }
    }
#pragma unroll
    for (int i = 0; i < 8; ++i) {
      int oo = o0 + i + 8 * ph;
      Out[((size_t)n * DD + oo) * TT + t0 + l15] = acc[i] + bias[oo];
    }
  }
}

// ---------------- layernorm over channel dim (per n,t column) + relu ----------------
__global__ __launch_bounds__(256)
void ln_ch_relu_kernel(float* __restrict__ A, const float* __restrict__ g,
                       const float* __restrict__ b)
{
  long w = (long)blockIdx.x * 8 + (threadIdx.x >> 5);  // one wave per (n,t)
  int lane = threadIdx.x & 31;
  long n = w / TT; int t = (int)(w % TT);
  float* base = A + (size_t)n * DD * TT + t;
  float vals[4]; float s = 0.f, s2 = 0.f;
#pragma unroll
  for (int j = 0; j < 4; ++j) {
    float v = base[(size_t)(lane + 32 * j) * TT];
    vals[j] = v; s += v; s2 += v * v;
  }
#pragma unroll
  for (int o = 1; o < 32; o <<= 1) { s += __shfl_xor(s, o, 32); s2 += __shfl_xor(s2, o, 32); }
  float mean = s * (1.f / DD);
  float inv  = rsqrtf(s2 * (1.f / DD) - mean * mean + LN_EPS);
#pragma unroll
  for (int j = 0; j < 4; ++j) {
    int d = lane + 32 * j;
    base[(size_t)d * TT] = fmaxf((vals[j] - mean) * inv * g[d] + b[d], 0.f);
  }
}

// ---------------- squeeze-excite + residual + relu ----------------
__global__ __launch_bounds__(128)
void se_kernel(const float* __restrict__ Out2, const float* __restrict__ Res,
               const float* __restrict__ w1, const float* __restrict__ b1,
               const float* __restrict__ w2, const float* __restrict__ b2,
               float* __restrict__ Dst)
{
  __shared__ float mean_s[DD], h1_s[DD / 4];
  int n = blockIdx.x, d = threadIdx.x;
  const float* row = Out2 + ((size_t)n * DD + d) * TT;
  float s = 0.f;
  for (int t = 0; t < TT; ++t) s += row[t];
  mean_s[d] = s * (1.f / TT);
  __syncthreads();
  if (d < 32) {
    float a = b1[d];
    for (int i = 0; i < DD; ++i) a += w1[d * DD + i] * mean_s[i];
    h1_s[d] = fmaxf(a, 0.f);
  }
  __syncthreads();
  float a = b2[d];
  for (int i = 0; i < 32; ++i) a += w2[d * 32 + i] * h1_s[i];
  float sg = 1.f / (1.f + __expf(-a));
  const float* rr = Res + ((size_t)n * DD + d) * TT;
  float* dr = Dst + ((size_t)n * DD + d) * TT;
  for (int t = 0; t < TT; ++t) dr[t] = fmaxf(row[t] * sg + rr[t], 0.f);
}

// ---------------- per-phase time sums (collapses strided pooling convs) ----------------
__global__ void phase_sum_kernel(const float* __restrict__ Hf, float* __restrict__ hbar,
                                 float* __restrict__ X1, float* __restrict__ X2)
{
  int i = blockIdx.x * blockDim.x + threadIdx.x;     // N*128
  if (i >= NN * DD) return;
  int d = i & 127; long n = i >> 7;
  const float* row = Hf + (size_t)i * TT;
  float s0 = 0, s1 = 0, s2 = 0, s3 = 0;
  for (int t = 0; t < TT; t += 4) { s0 += row[t]; s1 += row[t + 1]; s2 += row[t + 2]; s3 += row[t + 3]; }
  hbar[i] = (s0 + s1 + s2 + s3) * (1.f / TT);
  X1[n * 256 + d] = s1 + s3;            // odd t
  X1[n * 256 + 128 + d] = s0 + s2;      // even t
  float* x2 = X2 + n * 512;             // taps k=0..3 hit phases 2,3,0,1
  x2[d] = s2; x2[128 + d] = s3; x2[256 + d] = s0; x2[384 + d] = s1;
}

// ---------------- [O,I,Kc] fp32 -> [O, I*Kc] with k-major column blocks ----------------
__global__ void wcat_kernel(const float* __restrict__ w, float* __restrict__ out, int Kc)
{
  int i = blockIdx.x * blockDim.x + threadIdx.x;     // O*I*Kc
  if (i >= DD * DD * Kc) return;
  int k = i % Kc; int ii = (i / Kc) % DD; int o = i / (Kc * DD);
  out[(size_t)o * (DD * Kc) + k * DD + ii] = w[i];
}

// ---------------- row layernorm over last dim (=128), up to 2 residuals ----------------
__global__ __launch_bounds__(256)
void ln_row_kernel(const float* __restrict__ A, const float* __restrict__ R1,
                   const float* __restrict__ R2, const float* __restrict__ g,
                   const float* __restrict__ b, float* __restrict__ Out)
{
  long row = (long)blockIdx.x * 8 + (threadIdx.x >> 5);
  int lane = threadIdx.x & 31;
  const float* ar = A + row * DD;
  float vals[4]; float s = 0.f, s2 = 0.f;
#pragma unroll
  for (int j = 0; j < 4; ++j) {
    int d = lane + 32 * j;
    float v = ar[d];
    if (R1) v += R1[row * DD + d];
    if (R2) v += R2[row * DD + d];
    vals[j] = v; s += v; s2 += v * v;
  }
#pragma unroll
  for (int o = 1; o < 32; o <<= 1) { s += __shfl_xor(s, o, 32); s2 += __shfl_xor(s2, o, 32); }
  float mean = s * (1.f / DD);
  float inv  = rsqrtf(s2 * (1.f / DD) - mean * mean + LN_EPS);
#pragma unroll
  for (int j = 0; j < 4; ++j) {
    int d = lane + 32 * j;
    Out[row * DD + d] = (vals[j] - mean) * inv * g[d] + b[d];
  }
}

// ---------------- fills ----------------
__global__ void fill_f32(float* p, float v, long n) {
  long i = (long)blockIdx.x * blockDim.x + threadIdx.x; if (i < n) p[i] = v;
}
__global__ void fill_u32(unsigned* p, unsigned v, long n) {
  long i = (long)blockIdx.x * blockDim.x + threadIdx.x; if (i < n) p[i] = v;
}

// ---------------- graph attention (per-target softmax over incoming edges) ----------------
__device__ __forceinline__ unsigned f2mono(float f) {
  union { float f; unsigned u; } x; x.f = f;
  return (x.u & 0x80000000u) ? ~x.u : (x.u | 0x80000000u);
}
__device__ __forceinline__ float mono2f(unsigned m) {
  union { unsigned u; float f; } x;
  x.u = (m & 0x80000000u) ? (m & 0x7FFFFFFFu) : ~m;
  return x.f;
}

__global__ void edge_logits_kernel(const int* __restrict__ ei, const float* __restrict__ q,
                                   const float* __restrict__ k, float* __restrict__ logits,
                                   unsigned* __restrict__ umax)
{
  long i = (long)blockIdx.x * blockDim.x + threadIdx.x;   // E*H
  if (i >= (long)EE * HH) return;
  int h = (int)(i & 3); long e = i >> 2;
  int src = ei[e], dst = ei[EE + e];
  const float* qr = q + (size_t)dst * DD + h * HDIM;
  const float* kr = k + (size_t)src * DD + h * HDIM;
  float s = 0.f;
#pragma unroll
  for (int d = 0; d < HDIM; ++d) s += qr[d] * kr[d];
  s *= 0.17677669529f;                                    // 1/sqrt(32)
  logits[i] = s;
  atomicMax(&umax[(size_t)dst * HH + h], f2mono(s));
}

__global__ void edge_exp_kernel(const int* __restrict__ ei, float* __restrict__ logits,
                                const unsigned* __restrict__ umax, float* __restrict__ den)
{
  long i = (long)blockIdx.x * blockDim.x + threadIdx.x;
  if (i >= (long)EE * HH) return;
  int h = (int)(i & 3); long e = i >> 2;
  int dst = ei[EE + e];
  float ex = __expf(logits[i] - mono2f(umax[(size_t)dst * HH + h]));
  logits[i] = ex;
  atomicAdd(&den[(size_t)dst * HH + h], ex);
}

__global__ void edge_agg_kernel(const int* __restrict__ ei, const float* __restrict__ logits,
                                const float* __restrict__ den, const float* __restrict__ v,
                                float* __restrict__ agg)
{
  long i = (long)blockIdx.x * blockDim.x + threadIdx.x;
  if (i >= (long)EE * HH) return;
  int h = (int)(i & 3); long e = i >> 2;
  int src = ei[e], dst = ei[EE + e];
  float alpha = logits[i] / den[(size_t)dst * HH + h];
  const float* vr = v + (size_t)src * DD + h * HDIM;
  float* ar = agg + (size_t)dst * DD + h * HDIM;
#pragma unroll
  for (int d = 0; d < HDIM; ++d) atomicAdd(&ar[d], alpha * vr[d]);
}

// ---------------- flash attention over the 131072-slot memory bank ----------------
// grid = N/16 blocks, 128 threads (4 waves = 4 heads, one per wave).
// Kb: [M,128] bf16 row-major; VbT: [128,M] bf16 (transposed); Q,Out fp32 [N,128].
__global__ __launch_bounds__(128)
void mem_attn_kernel(const float* __restrict__ Q, const unsigned short* __restrict__ Kb,
                     const unsigned short* __restrict__ VbT, float* __restrict__ Out,
                     int Mtot)
{
  const int lane = threadIdx.x & 31;
  const int head = threadIdx.x >> 5;
  const int l15 = lane & 15, ph = lane >> 4;
  const long q0 = (long)blockIdx.x * 16;
  __shared__ __align__(16) unsigned short Pshared[HH][16][40];

  BF16x16 aq;                                           // Q tile as A (16q x 32d)
  {
    const float* qr = Q + (size_t)(q0 + l15) * DD + head * HDIM;
#pragma unroll
    for (int j = 0; j < 8; ++j) {
      aq.h[j]     = f2bf_bits(qr[ph * 8 + j]);
      aq.h[8 + j] = f2bf_bits(qr[16 + ph * 8 + j]);
    }
  }

  float mrun[8], lrun[8];
#pragma unroll
  for (int i = 0; i < 8; ++i) { mrun[i] = -1e30f; lrun[i] = 0.f; }
  v8f o0 = {}, o1 = {};
  const float sc = 0.17677669529f;

  for (int m0 = 0; m0 < Mtot; m0 += 32) {
    if (m0 + 32 < Mtot) {                               // stream next K/V chunk toward L2/L0
      __builtin_prefetch(Kb + (size_t)(m0 + 32 + l15) * DD + head * HDIM, 0, 1);
      __builtin_prefetch(VbT + (size_t)(head * HDIM + l15) * Mtot + m0 + 32, 0, 1);
    }
    // S = Q Kᵀ : two 16x16 tiles (mem cols 0-15, 16-31 of this chunk)
    BF16x16 b0, b1;
    const uint4* kr0 = (const uint4*)(Kb + (size_t)(m0 + l15) * DD + head * HDIM);
    const uint4* kr1 = (const uint4*)(Kb + (size_t)(m0 + 16 + l15) * DD + head * HDIM);
    b0.q[0] = kr0[ph]; b0.q[1] = kr0[2 + ph];
    b1.q[0] = kr1[ph]; b1.q[1] = kr1[2 + ph];
    v8f s0 = {}, s1 = {};
    s0 = wmma_bf16(aq, b0, s0);
    s1 = wmma_bf16(aq, b1, s1);

    // online softmax (row stats live in the lanes of each 16-lane half)
#pragma unroll
    for (int i = 0; i < 8; ++i) {
      float a = s0[i] * sc, b = s1[i] * sc;
      float mx = fmaxf(a, b);
#pragma unroll
      for (int o = 1; o < 16; o <<= 1) mx = fmaxf(mx, __shfl_xor(mx, o, 32));
      float mnew = fmaxf(mrun[i], mx);
      float corr = __expf(mrun[i] - mnew);
      float p0 = __expf(a - mnew), p1 = __expf(b - mnew);
      float psum = p0 + p1;
#pragma unroll
      for (int o = 1; o < 16; o <<= 1) psum += __shfl_xor(psum, o, 32);
      lrun[i] = lrun[i] * corr + psum;
      mrun[i] = mnew;
      o0[i] *= corr; o1[i] *= corr;
      Pshared[head][i + 8 * ph][l15]      = f2bf_bits(p0);   // C-layout -> LDS
      Pshared[head][i + 8 * ph][16 + l15] = f2bf_bits(p1);
    }

    BF16x16 ap;                                          // P back as A (16q x 32mem)
    {
      const unsigned short* pr = &Pshared[head][l15][0];
#pragma unroll
      for (int j = 0; j < 8; ++j) {
        ap.h[j]     = pr[ph * 8 + j];
        ap.h[8 + j] = pr[16 + ph * 8 + j];
      }
    }

    // O += P V : B tiles from VbT (dims 0-15 and 16-31 of this head)
    BF16x16 bv0, bv1;
    const uint4* vr0 = (const uint4*)(VbT + (size_t)(head * HDIM + l15) * Mtot + m0);
    const uint4* vr1 = (const uint4*)(VbT + (size_t)(head * HDIM + 16 + l15) * Mtot + m0);
    bv0.q[0] = vr0[ph]; bv0.q[1] = vr0[2 + ph];
    bv1.q[0] = vr1[ph]; bv1.q[1] = vr1[2 + ph];
    o0 = wmma_bf16(ap, bv0, o0);
    o1 = wmma_bf16(ap, bv1, o1);
  }

#pragma unroll
  for (int i = 0; i < 8; ++i) {
    float inv = 1.f / lrun[i];
    long qrow = q0 + i + 8 * ph;
    Out[qrow * DD + head * HDIM + l15]      = o0[i] * inv;
    Out[qrow * DD + head * HDIM + 16 + l15] = o1[i] * inv;
  }
}

// ---------------- final residual add ----------------
__global__ void add_kernel(const float* __restrict__ a, const float* __restrict__ b,
                           float* __restrict__ o, long n)
{
  long i = (long)blockIdx.x * blockDim.x + threadIdx.x;
  if (i < n) o[i] = a[i] + b[i];
}

// ============================= host launcher =============================
extern "C" void kernel_launch(void* const* d_in, const int* in_sizes, int n_in,
                              void* d_out, int out_size, void* d_ws, size_t ws_size,
                              hipStream_t stream)
{
  (void)in_sizes; (void)n_in; (void)out_size; (void)ws_size;
  auto F = [&](int i) { return (const float*)d_in[i]; };
  const float* x  = F(0);
  const float* ts = F(1);
  const int*   ei = (const int*)d_in[2];
  const float *te_freq = F(3), *te_phase = F(4);
  struct TB { const float *c1w,*c1b,*c2w,*c2b,*n1g,*n1b,*n2g,*n2b,*s1w,*s1b,*s2w,*s2b; };
  TB tb1{F(5),F(6),F(7),F(8),F(9),F(10),F(11),F(12),F(13),F(14),F(15),F(16)};
  TB tb2{F(17),F(18),F(19),F(20),F(21),F(22),F(23),F(24),F(25),F(26),F(27),F(28)};
  const float *pc0w=F(29), *pc0b=F(30), *pc1w=F(31), *pc1b=F(32), *pc2w=F(33), *pc2b=F(34);
  const float *fu1w=F(35), *fu1b=F(36), *fu2w=F(37), *fu2b=F(38), *png=F(39), *pnb=F(40);
  const float *gqw=F(41), *gqb=F(42), *gkw=F(43), *gkb=F(44), *gvw=F(45), *gvb=F(46);
  const float *gsw=F(47), *gsb=F(48), *gn1g=F(49), *gn1b=F(50), *gn2g=F(51), *gn2b=F(52);
  const float *ff1w=F(53), *ff1b=F(54), *ff2w=F(55), *ff2b=F(56);
  const float *memb=F(57);
  const float *mqw=F(58), *mqb=F(59), *mkw=F(60), *mkb=F(61), *mvw=F(62), *mvb=F(63);
  const float *mow=F(64), *mob=F(65);

  // ---- workspace carve ----
  const size_t NDT = (size_t)NN * DD * TT;
  char* base = (char*)d_ws; size_t ofs = 0;
  auto alloc = [&](size_t bytes) -> void* {
    void* p = base + ofs; ofs += (bytes + 255) & ~(size_t)255; return p;
  };
  float* h0 = (float*)alloc(NDT * 4);
  float* tA = (float*)alloc(NDT * 4);
  float* tB = (float*)alloc(NDT * 4);
  unsigned short* wc[4];
  for (int i = 0; i < 4; ++i) wc[i] = (unsigned short*)alloc((size_t)KC * DD * DD * 2);
  float* hbar = (float*)alloc((size_t)NN * DD * 4);
  float* X1   = (float*)alloc((size_t)NN * 256 * 4);
  float* X2   = (float*)alloc((size_t)NN * 512 * 4);
  float* W1c  = (float*)alloc((size_t)DD * 256 * 4);
  float* W2c  = (float*)alloc((size_t)DD * 512 * 4);
  float* fused   = (float*)alloc((size_t)NN * 384 * 4);
  float* fu1o    = (float*)alloc((size_t)NN * 256 * 4);
  float* pooledr = (float*)alloc((size_t)NN * DD * 4);
  float* pooled  = (float*)alloc((size_t)NN * DD * 4);
  float* qb  = (float*)alloc((size_t)NN * DD * 4);
  float* kb  = (float*)alloc((size_t)NN * DD * 4);
  float* vb  = (float*)alloc((size_t)NN * DD * 4);
  float* gso = (float*)alloc((size_t)NN * DD * 4);
  float* logits = (float*)alloc((size_t)EE * HH * 4);
  unsigned* umax = (unsigned*)alloc((size_t)NN * HH * 4);
  float* den = (float*)alloc((size_t)NN * HH * 4);
  float* agg = (float*)alloc((size_t)NN * DD * 4);
  float* g1   = (float*)alloc((size_t)NN * DD * 4);
  float* ffo  = (float*)alloc((size_t)NN * 512 * 4);
  float* ff2o = (float*)alloc((size_t)NN * DD * 4);
  float* gbuf = (float*)alloc((size_t)NN * DD * 4);
  float* Qm    = (float*)alloc((size_t)NN * DD * 4);
  float* attno = (float*)alloc((size_t)NN * DD * 4);
  float* reto  = (float*)alloc((size_t)NN * DD * 4);
  // tA is dead after the TCN: reuse it for the bf16 K/V memory bank (2x 33.5 MB)
  unsigned short* Kb  = (unsigned short*)tA;
  unsigned short* VbT = Kb + (size_t)MM * DD;

  // ---- 1. time encoding ----
  te_kernel<<<(int)(NDT / 256), 256, 0, stream>>>(x, ts, te_freq, te_phase, h0);

  // ---- 2. conv weight preps ----
  const float* cws[4] = { tb1.c1w, tb1.c2w, tb2.c1w, tb2.c2w };
  for (int i = 0; i < 4; ++i)
    convw_prep_kernel<<<(DD * DD * KC + 255) / 256, 256, 0, stream>>>(cws[i], wc[i]);

  // ---- 3. temporal block 1 (dil=1) ----
  conv_tcn_kernel<1><<<NN, 256, 0, stream>>>(h0, wc[0], tb1.c1b, tA);
  ln_ch_relu_kernel<<<NN * TT / 8, 256, 0, stream>>>(tA, tb1.n1g, tb1.n1b);
  conv_tcn_kernel<1><<<NN, 256, 0, stream>>>(tA, wc[1], tb1.c2b, tB);
  ln_ch_relu_kernel<<<NN * TT / 8, 256, 0, stream>>>(tB, tb1.n2g, tb1.n2b);
  se_kernel<<<NN, 128, 0, stream>>>(tB, h0, tb1.s1w, tb1.s1b, tb1.s2w, tb1.s2b, h0);

  // ---- 4. temporal block 2 (dil=2) ----
  conv_tcn_kernel<2><<<NN, 256, 0, stream>>>(h0, wc[2], tb2.c1b, tA);
  ln_ch_relu_kernel<<<NN * TT / 8, 256, 0, stream>>>(tA, tb2.n1g, tb2.n1b);
  conv_tcn_kernel<2><<<NN, 256, 0, stream>>>(tA, wc[3], tb2.c2b, tB);
  ln_ch_relu_kernel<<<NN * TT / 8, 256, 0, stream>>>(tB, tb2.n2g, tb2.n2b);
  se_kernel<<<NN, 128, 0, stream>>>(tB, h0, tb2.s1w, tb2.s1b, tb2.s2w, tb2.s2b, h0);

  // ---- 5. pooling collapsed to phase sums + three small GEMMs into fused[N,384] ----
  phase_sum_kernel<<<NN * DD / 256, 256, 0, stream>>>(h0, hbar, X1, X2);
  wcat_kernel<<<(DD * DD * 2 + 255) / 256, 256, 0, stream>>>(pc1w, W1c, 2);
  wcat_kernel<<<(DD * DD * 4 + 255) / 256, 256, 0, stream>>>(pc2w, W2c, 4);
  gemm_xwt_kernel<0,false,float><<<dim3(NN/16, 2), 128, 0, stream>>>(hbar, 128, pc0w, pc0b, fused, 384, 0,   128, 1.f);
  gemm_xwt_kernel<0,false,float><<<dim3(NN/16, 2), 128, 0, stream>>>(X1,   256, W1c,  pc1b, fused, 384, 128, 256, 1.f/33.f);
  gemm_xwt_kernel<0,false,float><<<dim3(NN/16, 2), 128, 0, stream>>>(X2,   512, W2c,  pc2b, fused, 384, 256, 512, 1.f/17.f);
  gemm_xwt_kernel<1,false,float><<<dim3(NN/16, 4), 128, 0, stream>>>(fused, 384, fu1w, fu1b, fu1o, 256, 0, 384, 1.f);
  gemm_xwt_kernel<0,false,float><<<dim3(NN/16, 2), 128, 0, stream>>>(fu1o, 256, fu2w, fu2b, pooledr, 128, 0, 256, 1.f);
  ln_row_kernel<<<NN / 8, 256, 0, stream>>>(pooledr, nullptr, nullptr, png, pnb, pooled);

  // ---- 6. graph transformer ----
  gemm_xwt_kernel<0,false,float><<<dim3(NN/16, 2), 128, 0, stream>>>(pooled, 128, gqw, gqb, qb,  128, 0, 128, 1.f);
  gemm_xwt_kernel<0,false,float><<<dim3(NN/16, 2), 128, 0, stream>>>(pooled, 128, gkw, gkb, kb,  128, 0, 128, 1.f);
  gemm_xwt_kernel<0,false,float><<<dim3(NN/16, 2), 128, 0, stream>>>(pooled, 128, gvw, gvb, vb,  128, 0, 128, 1.f);
  gemm_xwt_kernel<0,false,float><<<dim3(NN/16, 2), 128, 0, stream>>>(pooled, 128, gsw, gsb, gso, 128, 0, 128, 1.f);
  fill_u32<<<(NN * HH + 255) / 256, 256, 0, stream>>>(umax, 0u, (long)NN * HH);
  fill_f32<<<(NN * HH + 255) / 256, 256, 0, stream>>>(den, 0.f, (long)NN * HH);
  fill_f32<<<(NN * DD + 255) / 256, 256, 0, stream>>>(agg, 0.f, (long)NN * DD);
  edge_logits_kernel<<<(int)((long)EE * HH / 256), 256, 0, stream>>>(ei, qb, kb, logits, umax);
  edge_exp_kernel   <<<(int)((long)EE * HH / 256), 256, 0, stream>>>(ei, logits, umax, den);
  edge_agg_kernel   <<<(int)((long)EE * HH / 256), 256, 0, stream>>>(ei, logits, den, vb, agg);
  ln_row_kernel<<<NN / 8, 256, 0, stream>>>(agg, gso, pooled, gn1g, gn1b, g1);
  gemm_xwt_kernel<2,false,float><<<dim3(NN/16, 8), 128, 0, stream>>>(g1, 128, ff1w, ff1b, ffo, 512, 0, 128, 1.f);
  gemm_xwt_kernel<0,false,float><<<dim3(NN/16, 2), 128, 0, stream>>>(ffo, 512, ff2w, ff2b, ff2o, 128, 0, 512, 1.f);
  ln_row_kernel<<<NN / 8, 256, 0, stream>>>(ff2o, g1, nullptr, gn2g, gn2b, gbuf);

  // ---- 7. memory-bank attention (flash over M=131072, bf16 K/V resident in L2) ----
  gemm_xwt_kernel<0,false,unsigned short><<<dim3(MM/16, 2), 128, 0, stream>>>(memb, 128, mkw, mkb, Kb,  128, 0, 128, 1.f);
  gemm_xwt_kernel<0,true ,unsigned short><<<dim3(MM/16, 2), 128, 0, stream>>>(memb, 128, mvw, mvb, VbT, MM,  0, 128, 1.f);
  gemm_xwt_kernel<0,false,float><<<dim3(NN/16, 2), 128, 0, stream>>>(gbuf, 128, mqw, mqb, Qm, 128, 0, 128, 1.f);
  mem_attn_kernel<<<NN / 16, 128, 0, stream>>>(Qm, Kb, VbT, attno, MM);
  gemm_xwt_kernel<0,false,float><<<dim3(NN/16, 2), 128, 0, stream>>>(attno, 128, mow, mob, reto, 128, 0, 128, 1.f);

  // ---- 8. final residual ----
  add_kernel<<<(int)((long)NN * DD / 256), 256, 0, stream>>>(gbuf, reto, (float*)d_out, (long)NN * DD);
}
